// MCT_60284160966925
// MI455X (gfx1250) — matrix operand order repaired and checked by reading, not compiled, not run on hardware
//
#include <hip/hip_runtime.h>
#include <hip/hip_bf16.h>
#include <math.h>

typedef __attribute__((ext_vector_type(16))) _Float16 v16h;
typedef __attribute__((ext_vector_type(8)))  _Float16 v8h;
typedef __attribute__((ext_vector_type(8)))  float    v8f;

#define NTRK   192
#define DFEAT  4096
#define NPAIR  (NTRK * NTRK)

// ---------------------------------------------------------------------------
// helpers
// ---------------------------------------------------------------------------
static __device__ inline _Float16 habs16(_Float16 x) {
    unsigned short u = __builtin_bit_cast(unsigned short, x) & (unsigned short)0x7FFF;
    return __builtin_bit_cast(_Float16, u);
}

static __device__ inline v8f wmma16(v16h a, v16h b, v8f c) {
    return __builtin_amdgcn_wmma_f32_16x16x32_f16(false, a, false, b, (short)0, c,
                                                  false, false);
}

// A fragment, 16x32 f16 (M x K), A row-major with leading dim lda (halves).
// ISA 7.12.2: lanes 0-15 hold row M=lane, K={k0..k0+7} in v[0..7] halves,
// K={k0+16..k0+23} in v[8..15]; lanes 16-31 hold same rows with K offset +8.
static __device__ inline v16h load_a_f16(const _Float16* __restrict__ A, int lda,
                                         int m0, int k0, int lane) {
    int r  = lane & 15;
    int kk = k0 + ((lane >> 4) << 3);
    const _Float16* p = A + (size_t)(m0 + r) * (size_t)lda + kk;
    v8h lo = *(const v8h*)(p);
    v8h hi = *(const v8h*)(p + 16);
    v16h out;
#pragma unroll
    for (int t = 0; t < 8; ++t) { out[t] = lo[t]; out[8 + t] = hi[t]; }
    return out;
}

// Fused |g[i][k] - g[j0+r][k]| A fragment (the "dist" tensor, never materialized)
static __device__ inline v16h load_a_dist(const _Float16* __restrict__ g,
                                          int i, int j0, int k0, int lane) {
    int r  = lane & 15;
    int kk = k0 + ((lane >> 4) << 3);
    const _Float16* pj = g + (size_t)(j0 + r) * DFEAT + kk;
    const _Float16* pi = g + (size_t)i * DFEAT + kk;
    v8h jlo = *(const v8h*)(pj);
    v8h jhi = *(const v8h*)(pj + 16);
    v8h ilo = *(const v8h*)(pi);
    v8h ihi = *(const v8h*)(pi + 16);
    v16h out;
#pragma unroll
    for (int t = 0; t < 8; ++t) {
        out[t]     = habs16((_Float16)(ilo[t] - jlo[t]));
        out[8 + t] = habs16((_Float16)(ihi[t] - jhi[t]));
    }
    return out;
}

// B fragment, 32x16 f16 (K x N). B column n is row n of the torch-layout
// weight W[out][in] (ld = K halves). Lane L(0-15): col N=L, K={k0..k0+7} in v[0..7],
// K={k0+16..k0+23} in v[8..15]; lane L+16: same col with K offset +8.
static __device__ inline v16h load_b_w(const _Float16* __restrict__ W, int ldk,
                                       int n0, int k0, int lane) {
    int c  = lane & 15;
    int kk = k0 + ((lane >> 4) << 3);
    const _Float16* p = W + (size_t)(n0 + c) * (size_t)ldk + kk;
    v8h lo = *(const v8h*)(p);
    v8h hi = *(const v8h*)(p + 16);
    v16h out;
#pragma unroll
    for (int t = 0; t < 8; ++t) { out[t] = lo[t]; out[8 + t] = hi[t]; }
    return out;
}

// ---------------------------------------------------------------------------
// f32 -> f16 conversion (weights + input features)
// ---------------------------------------------------------------------------
__global__ void k_cvt_f16(const float* __restrict__ in, _Float16* __restrict__ out,
                          size_t n) {
    size_t i      = (size_t)blockIdx.x * blockDim.x + threadIdx.x;
    size_t stride = (size_t)gridDim.x * blockDim.x;
    for (; i < n; i += stride) out[i] = (_Float16)in[i];
}

// ---------------------------------------------------------------------------
// unified WMMA GEMM.  Wave tile: 32(M) x 64(N) = 8 accumulators; each B
// fragment is reused by two A fragments (12 b128 loads : 8 WMMAs per K-step).
// Block = 8 waves -> 256(M) x 64(N).
//   MODE 0: out = f16(relu(A @ W^T + bias))
//   MODE 1: out = f16(forig - relu(A @ W^T + bias))          (cam removal)
//   MODE 2: A rows are pairs p=i*192+j, A[p][k] = |g[i][k]-g[j][k]| (fused)
// ---------------------------------------------------------------------------
template <int MODE>
__global__ __launch_bounds__(256) void k_gemm(
        const _Float16* __restrict__ Asrc, const _Float16* __restrict__ Wh,
        const float* __restrict__ bias, const float* __restrict__ forig,
        _Float16* __restrict__ out, int M, int Nout, int K) {
    int lane = threadIdx.x & 31;
    int wave = threadIdx.x >> 5;
    int m0 = blockIdx.y * 256 + wave * 32;
    int n0 = blockIdx.x * 64;
    if (m0 >= M) return;

    int i = 0, j0 = 0;
    if (MODE == 2) { i = m0 / NTRK; j0 = m0 % NTRK; }  // 192 % 32 == 0

    v8f acc[8] = {};   // acc[mi*4 + t] : mi in {0,1} (M halves), t in 0..3 (N tiles)
    for (int k0 = 0; k0 < K; k0 += 32) {
        v16h a0, a1;
        if (MODE == 2) {
            a0 = load_a_dist(Asrc, i, j0,      k0, lane);
            a1 = load_a_dist(Asrc, i, j0 + 16, k0, lane);
        } else {
            a0 = load_a_f16(Asrc, K, m0,      k0, lane);
            a1 = load_a_f16(Asrc, K, m0 + 16, k0, lane);
        }
        v16h b0 = load_b_w(Wh, K, n0 +  0, k0, lane);
        v16h b1 = load_b_w(Wh, K, n0 + 16, k0, lane);
        v16h b2 = load_b_w(Wh, K, n0 + 32, k0, lane);
        v16h b3 = load_b_w(Wh, K, n0 + 48, k0, lane);
        acc[0] = wmma16(a0, b0, acc[0]);
        acc[1] = wmma16(a0, b1, acc[1]);
        acc[2] = wmma16(a0, b2, acc[2]);
        acc[3] = wmma16(a0, b3, acc[3]);
        acc[4] = wmma16(a1, b0, acc[4]);
        acc[5] = wmma16(a1, b1, acc[5]);
        acc[6] = wmma16(a1, b2, acc[6]);
        acc[7] = wmma16(a1, b3, acc[7]);
    }

    // epilogue: C/D layout — lane holds col N=lane&15, rows mb..mb+7
    int col = lane & 15;
    int mb  = (lane >> 4) << 3;
#pragma unroll
    for (int mi = 0; mi < 2; ++mi) {
#pragma unroll
        for (int t = 0; t < 4; ++t) {
            int n    = n0 + 16 * t + col;
            float bn = bias[n];
#pragma unroll
            for (int r = 0; r < 8; ++r) {
                size_t idx = (size_t)(m0 + mi * 16 + mb + r) * (size_t)Nout + n;
                float v    = acc[mi * 4 + t][r] + bn;
                float relu = v > 0.f ? v : 0.f;
                if (MODE == 1) out[idx] = (_Float16)(forig[idx] - relu);
                else           out[idx] = (_Float16)relu;
            }
        }
    }
}

// ---------------------------------------------------------------------------
// head: A[p] = sigmoid(h3[p] . Ws + bs); one wave per row (512 = 32 lanes x 16)
// ---------------------------------------------------------------------------
__global__ __launch_bounds__(256) void k_head(
        const _Float16* __restrict__ h3, const float* __restrict__ Ws,
        const float* __restrict__ bs, float* __restrict__ out, int M) {
    int row  = blockIdx.x * 8 + (threadIdx.x >> 5);
    int lane = threadIdx.x & 31;
    if (row >= M) return;
    const _Float16* p = h3 + (size_t)row * 512 + lane * 16;
    const float*    w = Ws + lane * 16;
    float s = 0.f;
#pragma unroll
    for (int t = 0; t < 16; ++t) s += (float)p[t] * w[t];
#pragma unroll
    for (int off = 16; off > 0; off >>= 1) s += __shfl_xor(s, off, 32);
    if (lane == 0) out[row] = 1.f / (1.f + __expf(-(s + bs[0])));
}

// ---------------------------------------------------------------------------
// launcher
// ---------------------------------------------------------------------------
extern "C" void kernel_launch(void* const* d_in, const int* in_sizes, int n_in,
                              void* d_out, int out_size, void* d_ws, size_t ws_size,
                              hipStream_t stream) {
    const float* f   = (const float*)d_in[0];
    const float* Wc1 = (const float*)d_in[1];
    const float* bc1 = (const float*)d_in[2];
    const float* W1  = (const float*)d_in[3];
    const float* b1  = (const float*)d_in[4];
    const float* W2  = (const float*)d_in[5];
    const float* b2  = (const float*)d_in[6];
    const float* W3  = (const float*)d_in[7];
    const float* b3  = (const float*)d_in[8];
    const float* Ws  = (const float*)d_in[9];
    const float* bs  = (const float*)d_in[10];
    float* out = (float*)d_out;

    // workspace carve-out (256B aligned)
    char*  base = (char*)d_ws;
    size_t off  = 0;
    auto alloc = [&](size_t bytes) -> void* {
        off = (off + 255) & ~(size_t)255;
        void* p = base + off;
        off += bytes;
        return p;
    };
    _Float16* fh   = (_Float16*)alloc((size_t)NTRK * DFEAT * 2);
    _Float16* gh   = (_Float16*)alloc((size_t)NTRK * DFEAT * 2);
    _Float16* Wc1h = (_Float16*)alloc((size_t)DFEAT * DFEAT * 2);
    _Float16* W1h  = (_Float16*)alloc((size_t)2048 * DFEAT * 2);
    _Float16* W2h  = (_Float16*)alloc((size_t)1024 * 2048 * 2);
    _Float16* W3h  = (_Float16*)alloc((size_t)512 * 1024 * 2);
    _Float16* h1   = (_Float16*)alloc((size_t)NPAIR * 2048 * 2);
    _Float16* h2   = (_Float16*)alloc((size_t)NPAIR * 1024 * 2);
    _Float16* h3   = (_Float16*)alloc((size_t)NPAIR * 512 * 2);
    (void)ws_size;

    // 1) f32 -> f16 conversions
    k_cvt_f16<<<1024, 256, 0, stream>>>(f,   fh,   (size_t)NTRK * DFEAT);
    k_cvt_f16<<<4096, 256, 0, stream>>>(Wc1, Wc1h, (size_t)DFEAT * DFEAT);
    k_cvt_f16<<<4096, 256, 0, stream>>>(W1,  W1h,  (size_t)2048 * DFEAT);
    k_cvt_f16<<<2048, 256, 0, stream>>>(W2,  W2h,  (size_t)1024 * 2048);
    k_cvt_f16<<<1024, 256, 0, stream>>>(W3,  W3h,  (size_t)512 * 1024);

    // 2) g = f - relu(f @ Wc1^T + bc1)         [192 x 4096]
    k_gemm<1><<<dim3(DFEAT / 64, (NTRK + 255) / 256), 256, 0, stream>>>(
        fh, Wc1h, bc1, f, gh, NTRK, DFEAT, DFEAT);

    // 3) h1 = relu(|g_i - g_j| @ W1^T + b1)    [36864 x 2048], dist fused
    k_gemm<2><<<dim3(2048 / 64, NPAIR / 256), 256, 0, stream>>>(
        gh, W1h, b1, nullptr, h1, NPAIR, 2048, DFEAT);

    // 4) h2 = relu(h1 @ W2^T + b2)             [36864 x 1024]
    k_gemm<0><<<dim3(1024 / 64, NPAIR / 256), 256, 0, stream>>>(
        h1, W2h, b2, nullptr, h2, NPAIR, 1024, 2048);

    // 5) h3 = relu(h2 @ W3^T + b3)             [36864 x 512]
    k_gemm<0><<<dim3(512 / 64, NPAIR / 256), 256, 0, stream>>>(
        h2, W3h, b3, nullptr, h3, NPAIR, 512, 1024);

    // 6) A = sigmoid(h3 @ Ws^T + bs)           [36864]
    k_head<<<(NPAIR + 7) / 8, 256, 0, stream>>>(h3, Ws, bs, out, NPAIR);
}
